// GraphTransformerNetwork_12893491822687
// MI455X (gfx1250) — matrix operand order repaired and checked by reading, not compiled, not run on hardware
//
#include <hip/hip_runtime.h>
#include <hip/hip_bf16.h>

#define N_NODES 50000
#define N_EDGES 600000
#define CH      128
#define OUT_CH  16
#define N_GRAPHS 512
#define LDW 144    // padded LDS row stride (elements); multiple of 16 keeps 32B alignment
#define ROWS 80    // rows per block: 50000/80 = 625 exact

typedef float  v8f   __attribute__((ext_vector_type(8)));
typedef __bf16 v16bf __attribute__((ext_vector_type(16)));
typedef __bf16 v8bf  __attribute__((ext_vector_type(8)));
typedef __bf16 v4bf  __attribute__((ext_vector_type(4)));

// ---------------- degree / normalization ----------------
__global__ void k_deg_init(float* deg) {
    int i = blockIdx.x * blockDim.x + threadIdx.x;
    if (i < N_NODES) deg[i] = 1.0f;   // self-loop
}

__global__ void k_deg_edges(const int* __restrict__ dst, float* deg) {
    int e = blockIdx.x * blockDim.x + threadIdx.x;
    if (e < N_EDGES) unsafeAtomicAdd(&deg[dst[e]], 1.0f);
}

__global__ void k_dinv(float* deg) {
    int i = blockIdx.x * blockDim.x + threadIdx.x;
    if (i < N_NODES) deg[i] = __frsqrt_rn(deg[i]);   // deg >= 1 always
}

// ---------------- WMMA GEMM: h = act(in) @ W ; acc_init = h * dinv^2 ----------------
// block = 256 threads (8 waves); block tile = 80 rows x 128 cols.
// Each wave owns one 16-col stripe and 5 row-tiles (5 accumulators, B-frag reuse).
__global__ void k_gemm_wmma(const float* __restrict__ in, const float* __restrict__ W,
                            const float* __restrict__ bias, int act,
                            const float* __restrict__ dinv,
                            float* __restrict__ hout, float* __restrict__ accout) {
    __shared__ alignas(32) __bf16 sW[CH * LDW];     // W transposed: sW[n*LDW + k]  (36864 B)
    __shared__ alignas(32) __bf16 sA[ROWS * LDW];   // sA[m*LDW + k]                (23040 B)

    const int tid  = threadIdx.x;
    const int row0 = blockIdx.x * ROWS;

    // stage W (transposed, bf16): 16384 floats as 4096 b128 loads
#pragma unroll 4
    for (int j = 0; j < 16; ++j) {
        int idx = (tid + j * 256) * 4;
        float4 w4 = *(const float4*)(W + idx);
        int k = idx >> 7, n = idx & 127;
        sW[(n + 0) * LDW + k] = (__bf16)w4.x;
        sW[(n + 1) * LDW + k] = (__bf16)w4.y;
        sW[(n + 2) * LDW + k] = (__bf16)w4.z;
        sW[(n + 3) * LDW + k] = (__bf16)w4.w;
    }
    // stage 80 input rows (activation fused for layer-2 input), b128 loads + b64 LDS stores
#pragma unroll 4
    for (int j = 0; j < (ROWS * CH) / 1024; ++j) {   // 10 iters
        int idx = (tid + j * 256) * 4;
        int m = idx >> 7, k = idx & 127;
        float4 a4 = *(const float4*)(in + (size_t)(row0 + m) * CH + k);
        float v0 = a4.x, v1 = a4.y, v2 = a4.z, v3 = a4.w;
        if (act) {
            v0 = fmaxf(v0 + bias[k + 0], 0.0f);
            v1 = fmaxf(v1 + bias[k + 1], 0.0f);
            v2 = fmaxf(v2 + bias[k + 2], 0.0f);
            v3 = fmaxf(v3 + bias[k + 3], 0.0f);
        }
        v4bf p = {(__bf16)v0, (__bf16)v1, (__bf16)v2, (__bf16)v3};
        *(v4bf*)(&sA[m * LDW + k]) = p;   // 8B aligned (k multiple of 4)
    }
    __syncthreads();

    const int wave = tid >> 5;
    const int lane = tid & 31;
    const int half = lane >> 4;
    const int l16  = lane & 15;

    const __bf16* bp = &sW[(wave * 16 + l16) * LDW];

    v8f c[5];
#pragma unroll
    for (int t = 0; t < 5; ++t)
        c[t] = (v8f){0.f, 0.f, 0.f, 0.f, 0.f, 0.f, 0.f, 0.f};

#pragma unroll
    for (int kk = 0; kk < CH; kk += 32) {
        // B 32x16 bf16: lane = column, 16 contiguous K at kk+half*16 (reused 5x)
        v16bf b = *(const v16bf*)(bp + kk + half * 16);
#pragma unroll
        for (int t = 0; t < 5; ++t) {
            const __bf16* ap = &sA[(t * 16 + l16) * LDW];
            // A 16x32 bf16: lane holds 16 contiguous K at kk+half*16 (matches B)
            v16bf a  = *(const v16bf*)(ap + kk + half * 16);
            c[t] = __builtin_amdgcn_wmma_f32_16x16x32_bf16(false, a, false, b, (short)0, c[t], false, false);
        }
    }

    const int col = wave * 16 + l16;
#pragma unroll
    for (int t = 0; t < 5; ++t) {
#pragma unroll
        for (int r = 0; r < 8; ++r) {
            int row  = row0 + t * 16 + r + half * 8;   // C/D layout: VGPR r -> M = r + 8*half
            float v  = c[t][r];
            float dv = dinv[row];
            hout[row * CH + col]   = v;
            accout[row * CH + col] = v * dv * dv;      // self-loop contribution (acc init)
        }
    }
}

// ---------------- edge scatter: acc[dst] += h[src] * dinv[src]*dinv[dst] ----------------
__global__ void k_scatter(const float* __restrict__ h, const int* __restrict__ src,
                          const int* __restrict__ dst, const float* __restrict__ dinv,
                          float* __restrict__ acc) {
    int gid = blockIdx.x * 256 + threadIdx.x;   // N_EDGES*128 threads total
    int e = gid >> 7, cidx = gid & 127;
    int s = src[e], d = dst[e];
    float w = dinv[s] * dinv[d];
    float v = h[s * CH + cidx] * w;
    unsafeAtomicAdd(&acc[d * CH + cidx], v);
}

// ---------------- pooling ----------------
__global__ void k_zero_pool(float* psum, float* pcnt) {
    int i = blockIdx.x * blockDim.x + threadIdx.x;
    if (i < N_GRAPHS * CH) psum[i] = 0.0f;
    if (i < N_GRAPHS) pcnt[i] = 0.0f;
}

__global__ void k_pool(const float* __restrict__ acc, const float* __restrict__ bias,
                       const int* __restrict__ batch,
                       float* __restrict__ psum, float* __restrict__ pcnt) {
    int gid = blockIdx.x * 256 + threadIdx.x;   // N_NODES*128 threads total
    int n = gid >> 7, cidx = gid & 127;
    float v = fmaxf(acc[n * CH + cidx] + bias[cidx], 0.0f);   // relu(gcn2 + b2)
    int g = batch[n];
    unsafeAtomicAdd(&psum[g * CH + cidx], v);
    if (cidx == 0) unsafeAtomicAdd(&pcnt[g], 1.0f);
}

// ---------------- head: out = sigmoid(mean @ Wfc + bfc) ----------------
__global__ void k_head(const float* __restrict__ psum, const float* __restrict__ pcnt,
                       const float* __restrict__ Wfc, const float* __restrict__ bfc,
                       float* __restrict__ out) {
    int gid = blockIdx.x * blockDim.x + threadIdx.x;   // N_GRAPHS*OUT_CH
    if (gid >= N_GRAPHS * OUT_CH) return;
    int g = gid >> 4, o = gid & 15;
    float inv = 1.0f / fmaxf(pcnt[g], 1.0f);
    float s = 0.0f;
#pragma unroll 8
    for (int k = 0; k < CH; ++k)
        s += psum[g * CH + k] * Wfc[k * OUT_CH + o];
    s = s * inv + bfc[o];
    out[gid] = 1.0f / (1.0f + __expf(-s));
}

extern "C" void kernel_launch(void* const* d_in, const int* in_sizes, int n_in,
                              void* d_out, int out_size, void* d_ws, size_t ws_size,
                              hipStream_t stream) {
    const float* x    = (const float*)d_in[0];
    const int*   ei   = (const int*)d_in[1];
    const int*   srcE = ei;
    const int*   dstE = ei + N_EDGES;
    const int*   batch= (const int*)d_in[2];
    const float* W1   = (const float*)d_in[3];
    const float* b1   = (const float*)d_in[4];
    const float* W2   = (const float*)d_in[5];
    const float* b2   = (const float*)d_in[6];
    const float* Wfc  = (const float*)d_in[7];
    const float* bfc  = (const float*)d_in[8];
    float* out = (float*)d_out;

    float* ws   = (float*)d_ws;
    float* dinv = ws;                                   // 50000 floats (64B aligned end)
    float* hbuf = dinv + N_NODES;                       // 6.4M floats
    float* accA = hbuf + (size_t)N_NODES * CH;          // 6.4M
    float* accB = accA + (size_t)N_NODES * CH;          // 6.4M
    float* psum = accB + (size_t)N_NODES * CH;          // 65536
    float* pcnt = psum + N_GRAPHS * CH;                 // 512

    // degrees + symmetric normalization
    k_deg_init <<<(N_NODES + 255) / 256, 256, 0, stream>>>(dinv);
    k_deg_edges<<<(N_EDGES + 255) / 256, 256, 0, stream>>>(dstE, dinv);
    k_dinv     <<<(N_NODES + 255) / 256, 256, 0, stream>>>(dinv);

    const int gemmBlocks = N_NODES / ROWS;          // 625, exact
    const int scatBlocks = (N_EDGES * CH) / 256;    // 300000, exact

    // layer 1
    k_gemm_wmma<<<gemmBlocks, 256, 0, stream>>>(x, W1, b1, 0, dinv, hbuf, accA);
    k_scatter  <<<scatBlocks, 256, 0, stream>>>(hbuf, srcE, dstE, dinv, accA);

    // layer 2 (relu(acc1 + b1) fused into GEMM A-load)
    k_gemm_wmma<<<gemmBlocks, 256, 0, stream>>>(accA, W2, b1, 1, dinv, hbuf, accB);
    k_scatter  <<<scatBlocks, 256, 0, stream>>>(hbuf, srcE, dstE, dinv, accB);

    // mean pool (relu(acc2 + b2) fused) + head
    k_zero_pool<<<(N_GRAPHS * CH + 255) / 256, 256, 0, stream>>>(psum, pcnt);
    k_pool     <<<(N_NODES * CH) / 256, 256, 0, stream>>>(accB, b2, batch, psum, pcnt);
    k_head     <<<(N_GRAPHS * OUT_CH + 255) / 256, 256, 0, stream>>>(psum, pcnt, Wfc, bfc, out);
}
